// VAE_23390391894450
// MI455X (gfx1250) — compile-verified, matrix operand order checked by reading, not run on hardware
//
#include <hip/hip_runtime.h>
#include <math.h>

typedef __bf16 bf16;
typedef __attribute__((ext_vector_type(16))) __bf16 bf16x16;
typedef __attribute__((ext_vector_type(8)))  __bf16 bf16x8;
typedef __attribute__((ext_vector_type(8)))  float  f32x8;
typedef __attribute__((ext_vector_type(4))) unsigned int u32x4;
typedef __attribute__((ext_vector_type(8))) int i32x8;
typedef __attribute__((ext_vector_type(4))) int i32x4;

// ---------- device helpers ----------

__device__ __forceinline__ f32x8 fzero() {
  f32x8 r;
#pragma unroll
  for (int i = 0; i < 8; ++i) r[i] = 0.f;
  return r;
}

// Load a 16x32 (rows x K) bf16 fragment. Works for both A (rows=batch) and B
// (rows = output unit of W[out][in], K = in). Per CDNA5 ISA layout:
// lane l holds row (l&15); element e -> K = (e/8)*16 + ((l>=16)?8:0) + (e%8).
__device__ __forceinline__ bf16x16 ldfrag(const bf16* base, int stride) {
  int lane = threadIdx.x & 31;
  const bf16* p = base + (lane & 15) * stride + ((lane >> 4) & 1) * 8;
  bf16x8 lo = *(const bf16x8*)(p);        // K chunk +0..7
  bf16x8 hi = *(const bf16x8*)(p + 16);   // K chunk +16..23
  bf16x16 f;
#pragma unroll
  for (int i = 0; i < 8; ++i) { f[i] = lo[i]; f[i + 8] = hi[i]; }
  return f;
}

__device__ __forceinline__ f32x8 wmma_bf(f32x8 acc, bf16x16 a, bf16x16 b) {
  return __builtin_amdgcn_wmma_f32_16x16x32_bf16(false, a, false, b, (short)0,
                                                 acc, false, false);
}

__device__ __forceinline__ float sigm(float x) { return 1.f / (1.f + __expf(-x)); }
__device__ __forceinline__ float gumbel(float u) {
  u = fminf(fmaxf(u, 1e-7f), 1.f - 1e-7f);
  return -logf(-logf(u));
}

// LDS byte offset of a pointer into a __shared__ array (generic -> AS3 -> int)
typedef __attribute__((address_space(3))) unsigned char lds_char;
__device__ __forceinline__ unsigned lds_addr_of(const void* p) {
  return (unsigned)(size_t)(lds_char*)p;
}

// Tensor Data Mover: DMA a 16-row x 2048-byte slab (16 rows of a [4096][1024]
// bf16 matrix) from global into LDS.  D# per CDNA5 ISA ch.8:
//  group0: count=1 | lds_addr | global_addr | type=2
//  group1: data_size=8B, tensor_dim0=256 units, tensor_dim1=4096 rows,
//          tile_dim0=256, tile_dim1=16, tensor_dim0_stride=256
__device__ __forceinline__ void tdm_load_16x1024(const bf16* gsrc,
                                                 unsigned lds_addr) {
  unsigned long long ga = (unsigned long long)(size_t)gsrc;
  u32x4 g0 = {1u, lds_addr, (unsigned)(ga & 0xffffffffull),
              (unsigned)((ga >> 32) & 0x1ffffffull) | (2u << 30)};
  i32x8 g1 = {(int)0x30000u,       // [17:16] data_size = 3 (8-byte units)
              (int)(256u << 16),   // tensor_dim0[15:0] @ bits[31:16]
              (int)(4096u << 16),  // dim0 hi16 = 0; tensor_dim1[15:0] @ [31:16]
              (int)(256u << 16),   // dim1 hi16 = 0; tile_dim0 @ [31:16]
              16,                  // tile_dim1 = 16 rows (tile_dim2 = 0)
              256,                 // tensor_dim0_stride (48b, low dword)
              0, 0};
  i32x4 g2 = {0, 0, 0, 0};
  i32x4 g3 = {0, 0, 0, 0};
#if __has_include(<hip/amd_detail/amd_gfx1250_TDM.h>)
  i32x8 g4 = {0, 0, 0, 0, 0, 0, 0, 0};
  __builtin_amdgcn_tensor_load_to_lds(g0, g1, g2, g3, g4, 0);
#else
  __builtin_amdgcn_tensor_load_to_lds(g0, g1, g2, g3, 0);
#endif
}

// ---------- one-time setup kernels ----------

__global__ void k_conv(const float* __restrict__ src, int srcStride, int Rin,
                       int Cin, int colOff, bf16* __restrict__ dst, int Cpad,
                       int total) {
  int idx = blockIdx.x * 256 + threadIdx.x;
  if (idx >= total) return;
  int r = idx / Cpad, c = idx - r * Cpad;
  float v = 0.f;
  if (r < Rin && c < Cin) v = src[r * srcStride + colOff + c];
  dst[idx] = (bf16)v;
}

// xp[b][0:160] = [prev0(5) | classifier(10) | z(128) | zero pad(17)]
__global__ void k_xp_init(const float* __restrict__ cls, const float* __restrict__ z,
                          bf16* __restrict__ xp) {
  int idx = blockIdx.x * 256 + threadIdx.x;
  if (idx >= 128 * 160) return;
  int b = idx / 160, c = idx - b * 160;
  float v = 0.f;
  if (c < 5)        v = (c == 2) ? 1.f : 0.f;
  else if (c < 15)  v = cls[b * 10 + (c - 5)];
  else if (c < 143) v = z[b * 128 + (c - 15)];
  xp[idx] = (bf16)v;
}

// s = tanh(z @ fc_in_W.T + b); scatter into h0(bf16), c0(f32), hh0(bf16), ch0(f32)
__global__ __launch_bounds__(256) void k_fcin(
    const bf16* __restrict__ zbf, const bf16* __restrict__ fcW,
    const float* __restrict__ fcb, bf16* __restrict__ h0, float* __restrict__ c0,
    bf16* __restrict__ hh0, float* __restrict__ ch0) {
  int wid = threadIdx.x >> 5, lane = threadIdx.x & 31;
  int rt = wid, ct = blockIdx.x;  // ct in [0,160): 2560 output cols
  f32x8 acc = fzero();
  for (int k = 0; k < 128; k += 32)
    acc = wmma_bf(acc, ldfrag(zbf + rt * 16 * 128 + k, 128),
                  ldfrag(fcW + (ct * 16) * 128 + k, 128));
  int hi8 = ((lane >> 4) & 1) * 8, ccol = lane & 15;
  int j = ct * 16 + ccol;
#pragma unroll
  for (int v = 0; v < 8; ++v) {
    int b = rt * 16 + hi8 + v;
    float s = tanhf(acc[v] + fcb[j]);
    if (j < 1024)      h0[b * 1024 + j] = (bf16)s;
    else if (j < 2048) c0[b * 1024 + (j - 1024)] = s;
    else if (j < 2304) hh0[b * 256 + (j - 2048)] = (bf16)s;
    else               ch0[b * 256 + (j - 2304)] = s;
  }
}

// ---------- per-step kernels ----------

// Hyper-LSTM: g4 = xp@WXx.T + hprev@WXh.T + hhprev@WHH.T + hb; update ch, hh
__global__ __launch_bounds__(128) void k_hyper(
    const bf16* __restrict__ xp, const bf16* __restrict__ hprev,
    const bf16* __restrict__ hhprev, const bf16* __restrict__ WXx,
    const bf16* __restrict__ WXh, const bf16* __restrict__ WHHb,
    const float* __restrict__ hb, float* __restrict__ ch,
    bf16* __restrict__ hhnext) {
  int wid = threadIdx.x >> 5, lane = threadIdx.x & 31;
  int tile = blockIdx.x * 4 + wid;  // 0..127 : 8 row-tiles x 16 col-tiles
  int rt = tile >> 4, ct = tile & 15;
  f32x8 acc[4] = {fzero(), fzero(), fzero(), fzero()};
  for (int k = 0; k < 160; k += 32) {
    bf16x16 a = ldfrag(xp + rt * 16 * 160 + k, 160);
#pragma unroll
    for (int g = 0; g < 4; ++g)
      acc[g] = wmma_bf(acc[g], a, ldfrag(WXx + (g * 256 + ct * 16) * 160 + k, 160));
  }
  for (int k = 0; k < 1024; k += 32) {
    bf16x16 a = ldfrag(hprev + rt * 16 * 1024 + k, 1024);
#pragma unroll
    for (int g = 0; g < 4; ++g)
      acc[g] = wmma_bf(acc[g], a, ldfrag(WXh + (g * 256 + ct * 16) * 1024 + k, 1024));
  }
  for (int k = 0; k < 256; k += 32) {
    bf16x16 a = ldfrag(hhprev + rt * 16 * 256 + k, 256);
#pragma unroll
    for (int g = 0; g < 4; ++g)
      acc[g] = wmma_bf(acc[g], a, ldfrag(WHHb + (g * 256 + ct * 16) * 256 + k, 256));
  }
  int hi8 = ((lane >> 4) & 1) * 8, ccol = lane & 15;
  int u = ct * 16 + ccol;
#pragma unroll
  for (int v = 0; v < 8; ++v) {
    int b = rt * 16 + hi8 + v;
    float gi = acc[0][v] + hb[u];
    float gf = acc[1][v] + hb[256 + u];
    float gg = acc[2][v] + hb[512 + u];
    float go = acc[3][v] + hb[768 + u];
    float chn = sigm(gf) * ch[b * 256 + u] + sigm(gi) * tanhf(gg);
    float hv = sigm(go) * tanhf(chn);
    ch[b * 256 + u] = chn;
    hhnext[b * 256 + u] = (bf16)hv;
  }
}

// zg_s = hh_n @ Whz_s.T for s in {x,h,b} (grid.y selects set), output bf16
__global__ __launch_bounds__(256) void k_zg(const bf16* __restrict__ hhn,
                                            const bf16* __restrict__ Whzall,
                                            bf16* __restrict__ zgall) {
  int wid = threadIdx.x >> 5, lane = threadIdx.x & 31;
  int tile = blockIdx.x * 8 + wid;  // 0..127
  int rt = tile >> 4, ct = tile & 15;
  const bf16* Wz = Whzall + (size_t)blockIdx.y * 256 * 256;
  bf16* zg = zgall + (size_t)blockIdx.y * 128 * 256;
  f32x8 acc = fzero();
  for (int k = 0; k < 256; k += 32)
    acc = wmma_bf(acc, ldfrag(hhn + rt * 16 * 256 + k, 256),
                  ldfrag(Wz + (ct * 16) * 256 + k, 256));
  int hi8 = ((lane >> 4) & 1) * 8, ccol = lane & 15;
#pragma unroll
  for (int v = 0; v < 8; ++v)
    zg[(rt * 16 + hi8 + v) * 256 + ct * 16 + ccol] = (bf16)acc[v];
}

// Main LSTM: pre = dX*gx + dH*gh + dB + b0 ; update c, write h_n (bf16).
// The per-gate Wh slab (16 rows x 1024 bf16 = 32KB), identical for all 8 waves
// of the block, is DMA'd into LDS by the Tensor Data Mover and double-buffered
// so gate g+1's DMA overlaps gate g's WMMAs.
__global__ __launch_bounds__(256) void k_main(
    const bf16* __restrict__ xp, const bf16* __restrict__ hprev,
    const bf16* __restrict__ zg,   // [3][128][256] bf16 (zg[b][g*64+z])
    const bf16* __restrict__ Wxp,  // [4096][160]
    const bf16* __restrict__ Whb,  // [4096][1024]
    const bf16* __restrict__ Wdz,  // [3][4][1024][64]
    const float* __restrict__ b0, float* __restrict__ c,
    bf16* __restrict__ hnext) {
  __shared__ bf16 sW[2][16 * 1024];  // double-buffered Wh gate slab
  int wid = threadIdx.x >> 5, lane = threadIdx.x & 31;
  int rt = wid, ct = blockIdx.x;  // ct in [0,64)
  int hi8 = ((lane >> 4) & 1) * 8, ccol = lane & 15;
  int hcol = ct * 16 + ccol;
  bool issuer = (threadIdx.x < 32);  // wave 0 drives the TDM

  if (issuer) {
    tdm_load_16x1024(Whb + (size_t)(0 * 1024 + ct * 16) * 1024,
                     lds_addr_of(&sW[0][0]));
    __builtin_amdgcn_s_wait_tensorcnt(0);
  }
  __syncthreads();

  f32x8 pre[4];
#pragma unroll
  for (int g = 0; g < 4; ++g) {
    if (g < 3 && issuer)  // prefetch next gate's slab into the other buffer
      tdm_load_16x1024(Whb + (size_t)((g + 1) * 1024 + ct * 16) * 1024,
                       lds_addr_of(&sW[(g + 1) & 1][0]));

    f32x8 agx = fzero();
    for (int k = 0; k < 160; k += 32)
      agx = wmma_bf(agx, ldfrag(xp + rt * 16 * 160 + k, 160),
                    ldfrag(Wxp + (g * 1024 + ct * 16) * 160 + k, 160));
    f32x8 agh = fzero();
    {
      const bf16* sB = &sW[g & 1][0];  // rows = Wh rows g*1024+ct*16 .. +15
      for (int k = 0; k < 1024; k += 32)
        agh = wmma_bf(agh, ldfrag(hprev + rt * 16 * 1024 + k, 1024),
                      ldfrag(sB + k, 1024));
    }
    f32x8 adx = fzero(), adh = fzero(), adb = fzero();
    for (int k = 0; k < 64; k += 32) {
      adx = wmma_bf(adx, ldfrag(zg + 0 * 32768 + rt * 16 * 256 + g * 64 + k, 256),
                    ldfrag(Wdz + (0 * 4096 + g * 1024 + ct * 16) * 64 + k, 64));
      adh = wmma_bf(adh, ldfrag(zg + 1 * 32768 + rt * 16 * 256 + g * 64 + k, 256),
                    ldfrag(Wdz + (1 * 4096 + g * 1024 + ct * 16) * 64 + k, 64));
      adb = wmma_bf(adb, ldfrag(zg + 2 * 32768 + rt * 16 * 256 + g * 64 + k, 256),
                    ldfrag(Wdz + (2 * 4096 + g * 1024 + ct * 16) * 64 + k, 64));
    }
    float bb = b0[g * 1024 + hcol];
#pragma unroll
    for (int v = 0; v < 8; ++v)
      pre[g][v] = adx[v] * agx[v] + adh[v] * agh[v] + adb[v] + bb;

    if (issuer) __builtin_amdgcn_s_wait_tensorcnt(0);
    __syncthreads();  // next buffer ready; all waves done with current buffer
  }
#pragma unroll
  for (int v = 0; v < 8; ++v) {
    int b = rt * 16 + hi8 + v;
    float cv = c[b * 1024 + hcol];
    float cn = sigm(pre[1][v]) * cv + sigm(pre[0][v]) * tanhf(pre[2][v]);
    float hv = sigm(pre[3][v]) * tanhf(cn);
    c[b * 1024 + hcol] = cn;
    hnext[b * 1024 + hcol] = (bf16)hv;
  }
}

// Projection + Gumbel sampling (single workgroup, y staged in LDS)
__global__ __launch_bounds__(256) void k_proj_sample(
    const bf16* __restrict__ hn, const bf16* __restrict__ pWb,
    const float* __restrict__ pb, const float* __restrict__ eps_t,
    const float* __restrict__ um_t, const float* __restrict__ up_t,
    float* __restrict__ out_t, bf16* __restrict__ xp) {
  __shared__ float y[128 * 128];
  int wid = threadIdx.x >> 5, lane = threadIdx.x & 31;
  int rt = wid;
  int hi8 = ((lane >> 4) & 1) * 8, ccol = lane & 15;
  for (int ct = 0; ct < 8; ++ct) {
    f32x8 acc = fzero();
    for (int k = 0; k < 1024; k += 32)
      acc = wmma_bf(acc, ldfrag(hn + rt * 16 * 1024 + k, 1024),
                    ldfrag(pWb + (ct * 16) * 1024 + k, 1024));
    int col = ct * 16 + ccol;
    float bias = (col < 123) ? pb[col] : 0.f;
#pragma unroll
    for (int v = 0; v < 8; ++v)
      y[(rt * 16 + hi8 + v) * 128 + col] = acc[v] + bias;
  }
  __syncthreads();
  if (threadIdx.x < 128) {
    int b = threadIdx.x;
    const float* yb = &y[b * 128];
    // argmax(log_softmax(pi)+g) == argmax(pi+g)
    float best = -3.4e38f; int kk = 0;
    for (int m = 0; m < 20; ++m) {
      float vv = yb[m] + gumbel(um_t[b * 20 + m]);
      if (vv > best) { best = vv; kk = m; }
    }
    float mx = yb[20 + kk], my = yb[40 + kk];
    float sx = __expf(yb[60 + kk]), sy = __expf(yb[80 + kk]);
    float rho = tanhf(yb[100 + kk]);
    float e0 = eps_t[b * 2 + 0], e1 = eps_t[b * 2 + 1];
    float dx = mx + sx * e0;
    float dy = my + sy * (rho * e0 + sqrtf(fmaxf(0.f, 1.f - rho * rho)) * e1);
    float pbest = -3.4e38f; int pidx = 0;
    for (int j = 0; j < 3; ++j) {
      float vv = yb[120 + j] + gumbel(up_t[b * 3 + j]);
      if (vv > pbest) { pbest = vv; pidx = j; }
    }
    float s0, s1, s2, s3, s4;
    if (pidx == 2) { s0 = 0.f; s1 = 0.f; s2 = 0.f; s3 = 0.f; s4 = 1.f; }
    else {
      s0 = dx; s1 = dy;
      s2 = (pidx == 0) ? 1.f : 0.f;
      s3 = (pidx == 1) ? 1.f : 0.f;
      s4 = 0.f;
    }
    out_t[b * 5 + 0] = s0; out_t[b * 5 + 1] = s1; out_t[b * 5 + 2] = s2;
    out_t[b * 5 + 3] = s3; out_t[b * 5 + 4] = s4;
    // next step's prev-stroke columns of xp
    xp[b * 160 + 0] = (bf16)s0; xp[b * 160 + 1] = (bf16)s1;
    xp[b * 160 + 2] = (bf16)s2; xp[b * 160 + 3] = (bf16)s3;
    xp[b * 160 + 4] = (bf16)s4;
  }
}

// ---------- host ----------

extern "C" void kernel_launch(void* const* d_in, const int* in_sizes, int n_in,
                              void* d_out, int out_size, void* d_ws, size_t ws_size,
                              hipStream_t stream) {
  (void)in_sizes; (void)n_in; (void)out_size; (void)ws_size;
  const float* z   = (const float*)d_in[0];
  const float* cls = (const float*)d_in[1];
  const float* fcW = (const float*)d_in[2];
  const float* fcb = (const float*)d_in[3];
  const float* Wx  = (const float*)d_in[4];
  const float* Wh  = (const float*)d_in[5];
  const float* b0  = (const float*)d_in[6];
  const float* hWx = (const float*)d_in[7];
  const float* hWh = (const float*)d_in[8];
  const float* hb  = (const float*)d_in[9];
  const float* Whz[3] = {(const float*)d_in[10], (const float*)d_in[11], (const float*)d_in[12]};
  const float* Wdz[3] = {(const float*)d_in[13], (const float*)d_in[14], (const float*)d_in[15]};
  const float* pW  = (const float*)d_in[16];
  const float* pb  = (const float*)d_in[17];
  const float* eps = (const float*)d_in[18];
  const float* um  = (const float*)d_in[19];
  const float* up  = (const float*)d_in[20];
  float* out = (float*)d_out;

  char* w = (char*)d_ws;
  size_t off = 0;
  auto abf = [&](size_t n) -> bf16* {
    bf16* p = (bf16*)(w + off); off += ((n * 2 + 255) & ~(size_t)255); return p;
  };
  auto af4 = [&](size_t n) -> float* {
    float* p = (float*)(w + off); off += ((n * 4 + 255) & ~(size_t)255); return p;
  };

  bf16* fcWb   = abf(2560 * 128);
  bf16* Wxp    = abf(4096 * 160);
  bf16* Whb    = abf(4096 * 1024);
  bf16* WXx    = abf(1024 * 160);
  bf16* WXh    = abf(1024 * 1024);
  bf16* WHHb   = abf(1024 * 256);
  bf16* Whzall = abf(3 * 256 * 256);
  bf16* Wdzall = abf(3 * 4096 * 64);
  bf16* pWb    = abf(128 * 1024);
  bf16* zbf    = abf(128 * 128);
  bf16* xp     = abf(128 * 160);
  bf16* hbf0   = abf(128 * 1024);
  bf16* hbf1   = abf(128 * 1024);
  bf16* hhbf0  = abf(128 * 256);
  bf16* hhbf1  = abf(128 * 256);
  bf16* zgall  = abf(3 * 128 * 256);
  float* c     = af4(128 * 1024);
  float* ch    = af4(128 * 256);

  auto conv = [&](const float* s, int ss, int Rin, int Cin, int co, bf16* d,
                  int Cp, int Rout) {
    int tot = Rout * Cp;
    k_conv<<<(tot + 255) / 256, 256, 0, stream>>>(s, ss, Rin, Cin, co, d, Cp, tot);
  };
  conv(fcW, 128, 2560, 128, 0, fcWb, 128, 2560);
  conv(Wx, 143, 4096, 143, 0, Wxp, 160, 4096);
  conv(Wh, 1024, 4096, 1024, 0, Whb, 1024, 4096);
  conv(hWx, 1167, 1024, 143, 0, WXx, 160, 1024);     // x part of hWx
  conv(hWx, 1167, 1024, 1024, 143, WXh, 1024, 1024); // h part of hWx
  conv(hWh, 256, 1024, 256, 0, WHHb, 256, 1024);
  for (int s = 0; s < 3; ++s)
    conv(Whz[s], 256, 256, 256, 0, Whzall + (size_t)s * 256 * 256, 256, 256);
  for (int s = 0; s < 3; ++s)
    conv(Wdz[s], 64, 4096, 64, 0, Wdzall + (size_t)s * 4096 * 64, 64, 4096);
  conv(pW, 1024, 123, 1024, 0, pWb, 1024, 128);      // pad rows 123..127 to 0
  conv(z, 128, 128, 128, 0, zbf, 128, 128);

  k_xp_init<<<(128 * 160 + 255) / 256, 256, 0, stream>>>(cls, z, xp);
  k_fcin<<<160, 256, 0, stream>>>(zbf, fcWb, fcb, hbf0, c, hhbf0, ch);

  for (int t = 0; t < 128; ++t) {
    bf16* hp  = (t & 1) ? hbf1 : hbf0;
    bf16* hn  = (t & 1) ? hbf0 : hbf1;
    bf16* hhp = (t & 1) ? hhbf1 : hhbf0;
    bf16* hhn = (t & 1) ? hhbf0 : hhbf1;
    k_hyper<<<32, 128, 0, stream>>>(xp, hp, hhp, WXx, WXh, WHHb, hb, ch, hhn);
    k_zg<<<dim3(16, 3), 256, 0, stream>>>(hhn, Whzall, zgall);
    k_main<<<64, 256, 0, stream>>>(xp, hp, zgall, Wxp, Whb, Wdzall, b0, c, hn);
    k_proj_sample<<<1, 256, 0, stream>>>(
        hn, pWb, pb, eps + (size_t)t * 128 * 2, um + (size_t)t * 128 * 20,
        up + (size_t)t * 128 * 3, out + (size_t)t * 128 * 5, xp);
  }
}
